// LearnableHash_10161892623056
// MI455X (gfx1250) — matrix operand order repaired and checked by reading, not compiled, not run on hardware
//
#include <hip/hip_runtime.h>
#include <hip/hip_bf16.h>
#include <math.h>

typedef _Float16 f16;
typedef __attribute__((ext_vector_type(16))) _Float16 v16h;
typedef __attribute__((ext_vector_type(8)))  _Float16 v8h;
typedef __attribute__((ext_vector_type(8)))  float    v8f;

#define RADIUS_F 1.3f
#define STEP_F 0.02f
#define NRAY 4096
#define NS   128
#define STOT (NRAY * NS)   // 524288

// ---------------- WMMA helpers (CDNA5 layouts) ----------------

__device__ __forceinline__ v8f wmma32(v16h a, v16h b, v8f c) {
  // D = A(16x32 f16) * B(32x16 f16) + C(16x16 f32)
  return __builtin_amdgcn_wmma_f32_16x16x32_f16(
      false, a, false, b, (short)0, c, false, false);
}

// A-fragment 16x32 f16: lane l -> row m = l&15, hi = l>>4.
// halves 0..15 = K[hi*16 .. hi*16+15] (contiguous).
// 'row' points at the 32-wide K row for this lane's m.
__device__ __forceinline__ v16h load_afrag(const f16* row, int hi) {
  v8h lo = *(const v8h*)(row + hi * 16);
  v8h hh = *(const v8h*)(row + hi * 16 + 8);
  v16h a;
#pragma unroll
  for (int t = 0; t < 8; ++t) { a[t] = lo[t]; a[t + 8] = hh[t]; }
  return a;
}

// B-fragment 32x16 f16 from transposed weights WT[N][K] (row-major):
// lane l -> col n = l&15 (+16*nt), halves h -> K = kc*32 + (l>>4)*16 + h.
// 16 contiguous halves -> one 32B load.
__device__ __forceinline__ v16h load_bfrag(const f16* WT, int K, int nt, int kc, int lane) {
  int n = nt * 16 + (lane & 15);
  int k = kc * 32 + ((lane >> 4) << 4);
  return *(const v16h*)(WT + (size_t)n * K + k);
}

// ---------------- Kernel P: transpose weights to f16 [N][K] ----------------

__global__ void k_prep(const float* __restrict__ W0, const float* __restrict__ W1,
                       const float* __restrict__ W2, const float* __restrict__ Wc0,
                       const float* __restrict__ Wc1, const float* __restrict__ Wc2,
                       f16* __restrict__ wt) {
  int tid = blockIdx.x * blockDim.x + threadIdx.x;
  int stride = gridDim.x * blockDim.x;
  f16* W0T  = wt;                 // [128][32]
  f16* W1T  = wt + 4096;          // [128][128]
  f16* W2T  = wt + 20480;         // [16][128]
  f16* Wc0T = wt + 22528;         // [64][32]  (K 31 -> pad to 32)
  f16* Wc1T = wt + 24576;         // [64][64]
  f16* Wc2T = wt + 28672;         // [16][64]  (N 3 -> pad to 16)
  for (int j = tid; j < 4096; j += stride)  { int n = j >> 5, k = j & 31;  W0T[j]  = (f16)W0[k * 128 + n]; }
  for (int j = tid; j < 16384; j += stride) { int n = j >> 7, k = j & 127; W1T[j]  = (f16)W1[k * 128 + n]; }
  for (int j = tid; j < 2048; j += stride)  { int n = j >> 7, k = j & 127; W2T[j]  = (f16)W2[k * 16 + n]; }
  for (int j = tid; j < 2048; j += stride)  { int n = j >> 5, k = j & 31;  Wc0T[j] = (k < 31) ? (f16)Wc0[k * 64 + n] : (f16)0.f; }
  for (int j = tid; j < 4096; j += stride)  { int n = j >> 6, k = j & 63;  Wc1T[j] = (f16)Wc1[k * 64 + n]; }
  for (int j = tid; j < 1024; j += stride)  { int n = j >> 6, k = j & 63;  Wc2T[j] = (n < 3) ? (f16)Wc2[k * 3 + n] : (f16)0.f; }
}

// ---------------- Kernel A: ray march + double trilinear sample ----------------

__device__ __forceinline__ void idxw(float c, int size, int& i0, int& i1, float& w) {
  float t = (c + 1.f) * 0.5f * (float)(size - 1);
  t = fminf(fmaxf(t, 0.f), (float)(size - 1));
  float f = floorf(t);
  i0 = (int)f;
  i1 = min(i0 + 1, size - 1);
  w = t - f;
}

template <int C>
__device__ __forceinline__ void trilin(const float* __restrict__ g, int size, int chanStride,
                                       float cx, float cy, float cz, float* out) {
  int x0, x1, y0, y1, z0, z1;
  float wx, wy, wz;
  idxw(cx, size, x0, x1, wx);
  idxw(cy, size, y0, y1, wy);
  idxw(cz, size, z0, z1, wz);
  int ys = size, zs = size * size;
  int b000 = z0 * zs + y0 * ys + x0, b001 = z0 * zs + y0 * ys + x1;
  int b010 = z0 * zs + y1 * ys + x0, b011 = z0 * zs + y1 * ys + x1;
  int b100 = z1 * zs + y0 * ys + x0, b101 = z1 * zs + y0 * ys + x1;
  int b110 = z1 * zs + y1 * ys + x0, b111 = z1 * zs + y1 * ys + x1;
  float w000 = (1.f - wx) * (1.f - wy) * (1.f - wz), w001 = wx * (1.f - wy) * (1.f - wz);
  float w010 = (1.f - wx) * wy * (1.f - wz),         w011 = wx * wy * (1.f - wz);
  float w100 = (1.f - wx) * (1.f - wy) * wz,         w101 = wx * (1.f - wy) * wz;
  float w110 = (1.f - wx) * wy * wz,                 w111 = wx * wy * wz;
#pragma unroll
  for (int c = 0; c < C; ++c) {
    const float* gc = g + (size_t)c * chanStride;
    out[c] = gc[b000] * w000 + gc[b001] * w001 + gc[b010] * w010 + gc[b011] * w011 +
             gc[b100] * w100 + gc[b101] * w101 + gc[b110] * w110 + gc[b111] * w111;
  }
}

__global__ __launch_bounds__(256) void k_sample(const float* __restrict__ ro, const float* __restrict__ rd,
                                                const float* __restrict__ G1, const float* __restrict__ F,
                                                f16* __restrict__ feats, unsigned char* __restrict__ maskb) {
  int s = blockIdx.x * blockDim.x + threadIdx.x;
  if (s >= STOT) return;
  int ray = s >> 7, i = s & 127;
  float ox = ro[ray * 3 + 0], oy = ro[ray * 3 + 1], oz = ro[ray * 3 + 2];
  float dx = rd[ray * 3 + 0], dy = rd[ray * 3 + 1], dz = rd[ray * 3 + 2];
  float inv = rsqrtf(dx * dx + dy * dy + dz * dz);
  dx *= inv; dy *= inv; dz *= inv;
  float b = ox * dx + oy * dy + oz * dz;
  float c = ox * ox + oy * oy + oz * oz - RADIUS_F * RADIUS_F;
  float disc = b * b - c;
  bool hit = disc > 0.f;
  float sq = sqrtf(hit ? disc : 1.f);
  float tmin = hit ? fmaxf(-b - sq, 0.f) : 0.f;
  float tmax = hit ? (-b + sq) : 0.f;
  float tmid = tmin + ((float)i + 0.5f) * STEP_F;
  bool mask = hit && (tmid <= tmax);
  float px = (ox + dx * tmid) / RADIUS_F;
  float py = (oy + dy * tmid) / RADIUS_F;
  float pz = (oz + dz * tmid) / RADIUS_F;

  float g1v[3];
  trilin<3>(G1, 256, 256 * 256 * 256, px, py, pz, g1v);
  float fv[32];
  trilin<32>(F, 16, 16 * 16 * 16, g1v[0], g1v[1], g1v[2], fv);

  f16* dst = feats + (size_t)s * 32;
#pragma unroll
  for (int j = 0; j < 32; ++j) dst[j] = (f16)fv[j];
  maskb[s] = mask ? 1 : 0;
}

// ---------------- Kernel B: fused sigma MLP (32->128->128->16), WMMA ----------------

__global__ __launch_bounds__(256) void k_sigma(const f16* __restrict__ feats,
                                               const f16* __restrict__ W0T, const f16* __restrict__ W1T,
                                               const f16* __restrict__ W2T,
                                               f16* __restrict__ sigOut, float* __restrict__ sigma,
                                               const unsigned char* __restrict__ maskb) {
  __shared__ f16 sH[8][16][136];  // per-wave 16x128 activation tile (padded rows)
  int lane = threadIdx.x & 31;
  int wid  = threadIdx.x >> 5;
  int s0   = blockIdx.x * 128 + wid * 16;
  int mrow = lane & 15, hi = lane >> 4;

  // Layer 0: A (16x32) straight from global feats; B from W0T[128][32]; N=128 (8 tiles).
  v16h a0 = load_afrag(feats + (size_t)(s0 + mrow) * 32, hi);
#pragma unroll
  for (int nt = 0; nt < 8; ++nt) {
    v8f acc = {};
    acc = wmma32(a0, load_bfrag(W0T, 32, nt, 0, lane), acc);
#pragma unroll
    for (int v = 0; v < 8; ++v)
      sH[wid][hi * 8 + v][nt * 16 + (lane & 15)] = (f16)fmaxf(acc[v], 0.f);
  }

  // Layer 1: K=128 (4 chunks), N=128. Load all A frags first, then overwrite sH in place.
  v16h A1[4];
#pragma unroll
  for (int kc = 0; kc < 4; ++kc)
    A1[kc] = load_afrag(&sH[wid][mrow][kc * 32], hi);
#pragma unroll
  for (int nt = 0; nt < 8; ++nt) {
    v8f acc = {};
#pragma unroll
    for (int kc = 0; kc < 4; ++kc)
      acc = wmma32(A1[kc], load_bfrag(W1T, 128, nt, kc, lane), acc);
#pragma unroll
    for (int v = 0; v < 8; ++v)
      sH[wid][hi * 8 + v][nt * 16 + (lane & 15)] = (f16)fmaxf(acc[v], 0.f);
  }

  // Layer 2: K=128 (4 chunks), N=16.
  v16h A2[4];
#pragma unroll
  for (int kc = 0; kc < 4; ++kc)
    A2[kc] = load_afrag(&sH[wid][mrow][kc * 32], hi);
  v8f acc = {};
#pragma unroll
  for (int kc = 0; kc < 4; ++kc)
    acc = wmma32(A2[kc], load_bfrag(W2T, 128, 0, kc, lane), acc);

  int n = lane & 15;
#pragma unroll
  for (int v = 0; v < 8; ++v) {
    int srow = s0 + hi * 8 + v;
    sigOut[(size_t)srow * 16 + n] = (f16)acc[v];
    if (n == 0)
      sigma[srow] = maskb[srow] ? fmaxf(acc[v], 0.f) : 0.f;
  }
}

// ---------------- Kernel C: SH encode + fused color MLP (32->64->64->3), WMMA ----------------

__global__ __launch_bounds__(256) void k_color(const f16* __restrict__ sigOut, const float* __restrict__ rd,
                                               const f16* __restrict__ Wc0T, const f16* __restrict__ Wc1T,
                                               const f16* __restrict__ Wc2T, float* __restrict__ col) {
  __shared__ f16 sX[8][16][40];   // per-wave 16x32 input tile
  __shared__ f16 sH[8][16][72];   // per-wave 16x64 hidden tile
  int lane = threadIdx.x & 31;
  int wid  = threadIdx.x >> 5;
  int s0   = blockIdx.x * 128 + wid * 16;
  int mrow = lane & 15, hi = lane >> 4;

  // lanes 0..15 each build one 32-wide input row: 16 SH coeffs + sig_out[1:16] + zero pad.
  if (lane < 16) {
    int s = s0 + lane;
    int ray = s >> 7;
    float dx = rd[ray * 3 + 0], dy = rd[ray * 3 + 1], dz = rd[ray * 3 + 2];
    float inv = rsqrtf(dx * dx + dy * dy + dz * dz);
    dx *= inv; dy *= inv; dz *= inv;
    float xx = dx * dx, yy = dy * dy, zz = dz * dz;
    float xy = dx * dy, yz = dy * dz, xz = dx * dz;
    float e[16];
    e[0]  = 0.28209479177387814f;
    e[1]  = -0.48860251190291987f * dy;
    e[2]  =  0.48860251190291987f * dz;
    e[3]  = -0.48860251190291987f * dx;
    e[4]  =  1.0925484305920792f * xy;
    e[5]  = -1.0925484305920792f * yz;
    e[6]  =  0.94617469575756f * zz - 0.31539156525252f;
    e[7]  = -1.0925484305920792f * xz;
    e[8]  =  0.5462742152960396f * (xx - yy);
    e[9]  =  0.5900435899266435f * dy * (-3.f * xx + yy);
    e[10] =  2.890611442640554f * xy * dz;
    e[11] =  0.4570457994644657f * dy * (1.f - 5.f * zz);
    e[12] =  0.3731763325901154f * dz * (5.f * zz - 3.f);
    e[13] =  0.4570457994644657f * dx * (1.f - 5.f * zz);
    e[14] =  1.445305721320277f * dz * (xx - yy);
    e[15] =  0.5900435899266435f * dx * (-xx + 3.f * yy);
    f16* row = &sX[wid][lane][0];
#pragma unroll
    for (int j = 0; j < 16; ++j) row[j] = (f16)e[j];
    const f16* so = sigOut + (size_t)s * 16;
#pragma unroll
    for (int j = 1; j < 16; ++j) row[15 + j] = so[j];
    row[31] = (f16)0.f;
  }
  // same-wave LDS ordering: no barrier needed.

  // Layer 0: K=32, N=64 (4 tiles).
  v16h a0 = load_afrag(&sX[wid][mrow][0], hi);
#pragma unroll
  for (int nt = 0; nt < 4; ++nt) {
    v8f acc = {};
    acc = wmma32(a0, load_bfrag(Wc0T, 32, nt, 0, lane), acc);
#pragma unroll
    for (int v = 0; v < 8; ++v)
      sH[wid][hi * 8 + v][nt * 16 + (lane & 15)] = (f16)fmaxf(acc[v], 0.f);
  }

  // Layer 1: K=64 (2 chunks), N=64.
  v16h A1[2];
#pragma unroll
  for (int kc = 0; kc < 2; ++kc)
    A1[kc] = load_afrag(&sH[wid][mrow][kc * 32], hi);
#pragma unroll
  for (int nt = 0; nt < 4; ++nt) {
    v8f acc = {};
#pragma unroll
    for (int kc = 0; kc < 2; ++kc)
      acc = wmma32(A1[kc], load_bfrag(Wc1T, 64, nt, kc, lane), acc);
#pragma unroll
    for (int v = 0; v < 8; ++v)
      sH[wid][hi * 8 + v][nt * 16 + (lane & 15)] = (f16)fmaxf(acc[v], 0.f);
  }

  // Layer 2: K=64 (2 chunks), N=16 (cols 0..2 valid).
  v16h A2[2];
#pragma unroll
  for (int kc = 0; kc < 2; ++kc)
    A2[kc] = load_afrag(&sH[wid][mrow][kc * 32], hi);
  v8f acc = {};
#pragma unroll
  for (int kc = 0; kc < 2; ++kc)
    acc = wmma32(A2[kc], load_bfrag(Wc2T, 64, 0, kc, lane), acc);

  int n = lane & 15;
  if (n < 3) {
#pragma unroll
    for (int v = 0; v < 8; ++v) {
      int srow = s0 + hi * 8 + v;
      col[(size_t)srow * 3 + n] = acc[v];
    }
  }
}

// ---------------- Kernel D: per-ray alpha compositing scan ----------------

__global__ __launch_bounds__(256) void k_comp(const float* __restrict__ sigma,
                                              const float* __restrict__ col,
                                              float* __restrict__ out) {
  int ray = blockIdx.x * blockDim.x + threadIdx.x;
  if (ray >= NRAY) return;
  float T = 1.f, wsum = 0.f, r = 0.f, g = 0.f, bl = 0.f;
  for (int i = 0; i < NS; ++i) {
    int s = ray * NS + i;
    float sg = sigma[s];
    float alpha = 1.f - __expf(-sg * STEP_F);
    float w = alpha * T;
    T *= (1.f - alpha + 1e-10f);
    const float* cp = col + (size_t)s * 3;
    r  += w / (1.f + __expf(-cp[0]));
    g  += w / (1.f + __expf(-cp[1]));
    bl += w / (1.f + __expf(-cp[2]));
    wsum += w;
  }
  float bg = 1.f - wsum;
  out[ray * 3 + 0] = r + bg;
  out[ray * 3 + 1] = g + bg;
  out[ray * 3 + 2] = bl + bg;
}

// ---------------- launcher ----------------

extern "C" void kernel_launch(void* const* d_in, const int* in_sizes, int n_in,
                              void* d_out, int out_size, void* d_ws, size_t ws_size,
                              hipStream_t stream) {
  const float* rays_o = (const float*)d_in[0];
  const float* rays_d = (const float*)d_in[1];
  const float* G1     = (const float*)d_in[2];
  const float* F      = (const float*)d_in[3];
  const float* W0     = (const float*)d_in[4];
  const float* W1     = (const float*)d_in[5];
  const float* W2     = (const float*)d_in[6];
  const float* Wc0    = (const float*)d_in[7];
  const float* Wc1    = (const float*)d_in[8];
  const float* Wc2    = (const float*)d_in[9];

  char* ws = (char*)d_ws;
  // workspace layout (bytes)
  f16*   wt     = (f16*)ws;                       // 29696 halves of transposed weights
  f16*   feats  = (f16*)(ws + 65536);             // [S][32] f16  = 32 MB
  f16*   sigOut = (f16*)(ws + 33619968);          // [S][16] f16  = 16 MB
  float* sigma  = (float*)(ws + 50397184);        // [S] f32      = 2 MB
  float* col    = (float*)(ws + 52494336);        // [S][3] f32   = 6 MB
  unsigned char* maskb = (unsigned char*)(ws + 58785792); // [S] u8

  float* out = (float*)d_out;

  k_prep<<<16, 256, 0, stream>>>(W0, W1, W2, Wc0, Wc1, Wc2, wt);
  k_sample<<<STOT / 256, 256, 0, stream>>>(rays_o, rays_d, G1, F, feats, maskb);
  k_sigma<<<STOT / 128, 256, 0, stream>>>(feats, wt, wt + 4096, wt + 20480,
                                          sigOut, sigma, maskb);
  k_color<<<STOT / 128, 256, 0, stream>>>(sigOut, rays_d, wt + 22528, wt + 24576,
                                          wt + 28672, col);
  k_comp<<<16, 256, 0, stream>>>(sigma, col, out);
}
